// ConvUnit_29368986370419
// MI455X (gfx1250) — 8-bit matrix operand order repaired and checked by reading, compiled, not run on hardware
//
#include <hip/hip_runtime.h>

typedef __attribute__((ext_vector_type(8))) int   v8i;
typedef __attribute__((ext_vector_type(8))) float v8f;

#define LW_DW (9*4*32*8)

__device__ __forceinline__ unsigned int f32_to_e4m3(float f) {
  unsigned int u    = __float_as_uint(f);
  unsigned int sign = (u >> 24) & 0x80u;
  unsigned int au   = u & 0x7FFFFFFFu;
  if (au >= 0x7F800000u) return sign | 0x7Fu;
  if (au >= 0x43E00000u) return sign | 0x7Eu;
  int exp32 = (int)(au >> 23);
  unsigned int mant = au & 0x7FFFFFu;
  int Et = exp32 - 120;
  unsigned int v;
  if (Et >= 1) {
    unsigned int top  = mant >> 20;
    unsigned int rest = mant & 0xFFFFFu;
    v = ((unsigned int)Et << 3) | top;
    v += (rest > 0x80000u) || (rest == 0x80000u && (v & 1u));
    if (v > 0x7Eu) v = 0x7Eu;
  } else {
    int shift = 21 - Et;
    if (shift > 31) shift = 31;
    unsigned int m   = mant | 0x800000u;
    v = m >> shift;
    unsigned int rem  = m & ((1u << shift) - 1u);
    unsigned int half = 1u << (shift - 1);
    v += (rem > half) || (rem == half && (v & 1u));
  }
  return sign | v;
}

__device__ __forceinline__ unsigned int weight_frag_dword(const float* __restrict__ w,
                                                          int idx) {
  const int j   = idx & 7;
  const int l   = (idx >> 3) & 31;
  const int nt  = (idx >> 8) & 3;
  const int tap = idx >> 10;
  const int hi  = j >> 2;
  const int c0  = (hi ? 32 : 0) + ((l & 16) ? 16 : 0) + (j & 3) * 4;
  const int o   = nt * 16 + (l & 15);
  const float* wp = w + (o * 64 + c0) * 9 + tap;
  return  f32_to_e4m3(wp[0])
       | (f32_to_e4m3(wp[9])  << 8)
       | (f32_to_e4m3(wp[18]) << 16)
       | (f32_to_e4m3(wp[27]) << 24);
}

__global__ __launch_bounds__(256)
void prep_weights_fp8(const float* __restrict__ w, unsigned int* __restrict__ wfrag) {
  const int idx = blockIdx.x * 256 + threadIdx.x;
  if (idx < LW_DW) wfrag[idx] = weight_frag_dword(w, idx);
}

template <bool USE_PREP>
__global__ __launch_bounds__(256)
void cimu_conv2d_bitserial_fp8_wmma(const float* __restrict__ x,
                                    const float* __restrict__ w,
                                    const unsigned int* __restrict__ wfrag,
                                    const float* __restrict__ bias,
                                    float* __restrict__ out)
{
  __shared__ unsigned int  LW[LW_DW];
  __shared__ unsigned char LP[6*18*72];
  __shared__ float         LO[64*68];

  const int tid  = threadIdx.x;
  const int lane = tid & 31;
  const int wave = tid >> 5;
  const int lhal = lane & 16;
  const int l15  = lane & 15;

  int bid = blockIdx.x;
  const int colTile = bid % 7;   bid /= 7;
  const int rowTile = bid % 28;  bid /= 28;
  const int batch   = bid;
  const int hr0 = rowTile * 4;
  const int wc0 = colTile * 16;

  if (USE_PREP) {
    const uint4* src = (const uint4*)wfrag;
    uint4*       dst = (uint4*)LW;
#pragma unroll
    for (int i = 0; i < LW_DW / 4 / 256; ++i)
      dst[tid + i * 256] = src[tid + i * 256];
  } else {
    for (int idx = tid; idx < LW_DW; idx += 256)
      LW[idx] = weight_frag_dword(w, idx);
  }

  for (int i = tid; i < 6*18*64; i += 256) {
    const int col = i % 18;
    const int row = (i / 18) % 6;
    const int c   = i / 108;
    int h  = hr0 + row; h  = h  > 111 ? 111 : h;
    int ww = wc0 + col; ww = ww > 111 ? 111 : ww;
    float v = x[((batch * 64 + c) * 112 + h) * 112 + ww];
    v = fminf(fmaxf(v, -128.0f), 127.0f);
    LP[(row * 18 + col) * 72 + c] = (unsigned char)(signed char)v;
  }

  for (int i = tid; i < 64*68; i += 256) LO[i] = 0.0f;

  __syncthreads();

  v8f acc[4][4];
  const v8f vzero = {0.f,0.f,0.f,0.f,0.f,0.f,0.f,0.f};
#pragma unroll
  for (int m = 0; m < 4; ++m)
#pragma unroll
    for (int n = 0; n < 4; ++n) acc[m][n] = vzero;

  const unsigned int bit = (unsigned int)wave;
  const unsigned int m1  = 0x01010101u << bit;
  const unsigned int lsh = (bit < 7u) ? (6u - bit) : 0u;
  const unsigned int rsh = (bit < 7u) ? 0u : 1u;
  const int chalf = lhal ? 16 : 0;

  for (int tap = 0; tap < 9; ++tap) {
    const int kh = tap / 3;
    const int kw = tap - kh * 3;

    v8i bfrag[4];
#pragma unroll
    for (int nt = 0; nt < 4; ++nt)
      bfrag[nt] = *(const v8i*)&LW[((tap * 4 + nt) * 32 + lane) * 8];

    unsigned int raw[4][8];
#pragma unroll
    for (int msub = 0; msub < 4; ++msub) {
      const unsigned char* p =
          &LP[((msub + kh) * 18 + (l15 + kw)) * 72 + chalf];
#pragma unroll
      for (int q = 0; q < 4; ++q) {
        raw[msub][2*q]   = *(const unsigned int*)(p + (q >> 1) * 32 + (q & 1) * 8);
        raw[msub][2*q+1] = *(const unsigned int*)(p + (q >> 1) * 32 + (q & 1) * 8 + 4);
      }
    }

#pragma unroll
    for (int msub = 0; msub < 4; ++msub) {
      v8i a;
#pragma unroll
      for (int q = 0; q < 8; ++q)
        a[q] = (int)(((raw[msub][q] & m1) << lsh) >> rsh);
#pragma unroll
      for (int nt = 0; nt < 4; ++nt) {
        acc[msub][nt] = __builtin_amdgcn_wmma_f32_16x16x64_fp8_fp8(
            a, bfrag[nt], (short)0, acc[msub][nt], false, false);
      }
    }
  }

  const float pwv = (wave == 7) ? -128.0f : (float)(1 << wave);
#pragma unroll
  for (int msub = 0; msub < 4; ++msub)
#pragma unroll
    for (int nt = 0; nt < 4; ++nt)
#pragma unroll
      for (int j = 0; j < 8; ++j) {
        float y2 = acc[msub][nt][j];
        float q = rintf(y2 * 0.03125f);
        q = fminf(fmaxf(q, -128.0f), 127.0f);
        const int m = msub * 16 + (lhal ? j + 8 : j);
        const int o = nt * 16 + l15;
        atomicAdd(&LO[m * 68 + o], pwv * (q * 16.0f));
      }

  __syncthreads();

  for (int i = tid; i < 64 * 64; i += 256) {
    const int m  = i & 63;
    const int o  = i >> 6;
    const int oh = hr0 + (m >> 4);
    const int ow = wc0 + (m & 15);
    if (oh < 110 && ow < 110)
      out[((batch * 64 + o) * 110 + oh) * 110 + ow] = LO[m * 68 + o] + bias[o];
  }
}

extern "C" void kernel_launch(void* const* d_in, const int* in_sizes, int n_in,
                              void* d_out, int out_size, void* d_ws, size_t ws_size,
                              hipStream_t stream) {
  (void)in_sizes; (void)n_in; (void)out_size;
  const float* x  = (const float*)d_in[0];
  const float* w  = (const float*)d_in[1];
  const float* b  = (const float*)d_in[2];
  float* out = (float*)d_out;
  dim3 grid(16 * 28 * 7);
  if (ws_size >= (size_t)(LW_DW * 4)) {
    prep_weights_fp8<<<dim3((LW_DW + 255) / 256), dim3(256), 0, stream>>>(
        w, (unsigned int*)d_ws);
    cimu_conv2d_bitserial_fp8_wmma<true><<<grid, dim3(256), 0, stream>>>(
        x, w, (const unsigned int*)d_ws, b, out);
  } else {
    cimu_conv2d_bitserial_fp8_wmma<false><<<grid, dim3(256), 0, stream>>>(
        x, w, nullptr, b, out);
  }
}
